// BERT_SCL_62130996904570
// MI455X (gfx1250) — matrix operand order repaired and checked by reading, not compiled, not run on hardware
//
#include <hip/hip_runtime.h>
#include <hip/hip_bf16.h>
#include <math.h>

// MI455X (gfx1250) wave32 implementation.
// Heavy op: 8192x8192 Gram matrix over K=768 => 103 GFLOP via
// v_wmma_f32_16x16x32_f16. en (f16) is 12.6MB -> L2 resident; compute bound.
// Streaming softmax with fixed shift 1/TEMP (S <= 1/TEMP always).
// Round 2: branchless epilogue (v_cndmask instead of EXEC branches),
// dual column tiles per A fragment (halves ds_load traffic per WMMA).

typedef __attribute__((ext_vector_type(16))) _Float16 v16h;
typedef __attribute__((ext_vector_type(8)))  _Float16 v8h;
typedef __attribute__((ext_vector_type(8)))  float    v8f;

#define B_ROWS 8192
#define D_DIM  768
#define NLAB   7
#define TEMPR  (1.0f / 0.3f)   /* 1/TEMP, also the logsumexp shift (max possible S) */
#define LAMBDA 0.9f

// ---------------------------------------------------------------- kernel 1
// Row-normalize cls_emb and emit f16 copy (row-major) for the WMMA kernel.
__global__ __launch_bounds__(256) void normalize_f16_kernel(
    const float* __restrict__ cls, _Float16* __restrict__ enh) {
  __shared__ float wred[8];
  __shared__ float stot;
  const int r = blockIdx.x;
  const float* row = cls + (size_t)r * D_DIM;
  const int t = threadIdx.x;
  float x0 = row[t], x1 = row[t + 256], x2 = row[t + 512];
  float p = x0 * x0 + x1 * x1 + x2 * x2;
  for (int m = 16; m >= 1; m >>= 1) p += __shfl_xor(p, m, 32);
  if ((t & 31) == 0) wred[t >> 5] = p;
  __syncthreads();
  if (t == 0) {
    float s = 0.f;
    for (int i = 0; i < 8; i++) s += wred[i];
    stot = s;
  }
  __syncthreads();
  const float inv = 1.0f / sqrtf(stot);
  _Float16* orow = enh + (size_t)r * D_DIM;
  orow[t]       = (_Float16)(x0 * inv);
  orow[t + 256] = (_Float16)(x1 * inv);
  orow[t + 512] = (_Float16)(x2 * inv);
}

// ---------------------------------------------------------------- kernel 2
// logits = pooled @ W^T + b  (N=7, memory bound) + per-row CE term.
__global__ __launch_bounds__(256) void logits_ce_kernel(
    const float* __restrict__ pooled, const float* __restrict__ W,
    const float* __restrict__ bias, const int* __restrict__ labels,
    float* __restrict__ logits_out, float* __restrict__ ce_row) {
  __shared__ float wred[8][NLAB];
  __shared__ float lg[NLAB];
  const int r = blockIdx.x;
  const int t = threadIdx.x;
  const float* prow = pooled + (size_t)r * D_DIM;
  float acc[NLAB];
#pragma unroll
  for (int c = 0; c < NLAB; c++) acc[c] = 0.f;
  for (int i = t; i < D_DIM; i += 256) {
    const float pv = prow[i];
#pragma unroll
    for (int c = 0; c < NLAB; c++) acc[c] += pv * W[c * D_DIM + i];
  }
#pragma unroll
  for (int c = 0; c < NLAB; c++)
    for (int m = 16; m >= 1; m >>= 1) acc[c] += __shfl_xor(acc[c], m, 32);
  if ((t & 31) == 0)
#pragma unroll
    for (int c = 0; c < NLAB; c++) wred[t >> 5][c] = acc[c];
  __syncthreads();
  if (t < NLAB) {
    float s = bias[t];
    for (int w = 0; w < 8; w++) s += wred[w][t];
    lg[t] = s;
    logits_out[(size_t)r * NLAB + t] = s;
  }
  __syncthreads();
  if (t == 0) {
    float mx = lg[0];
#pragma unroll
    for (int c = 1; c < NLAB; c++) mx = fmaxf(mx, lg[c]);
    float se = 0.f;
#pragma unroll
    for (int c = 0; c < NLAB; c++) se += __expf(lg[c] - mx);
    const float lse = mx + __logf(se);
    ce_row[r] = lse - lg[labels[r]];
  }
}

// ---------------------------------------------------------------- kernel 3
// Streaming Gram matrix + supervised-contrastive per-row term.
// Block = 8 waves, owns a 16-row strip (A tile in LDS, reused 8192x).
// Each wave sweeps 1024 columns as 32 pairs of 16-col tiles; K=768 in
// 24 WMMA steps per tile; one A fragment feeds two WMMAs (two B tiles).
__global__ __launch_bounds__(256) void scl_gram_kernel(
    const _Float16* __restrict__ enh, const int* __restrict__ labels,
    float* __restrict__ contr_row) {
  __shared__ _Float16 As[16 * D_DIM];      // 24KB f16 A strip
  __shared__ int   llab[16];
  __shared__ float s_sumexp[16], s_possum[16], s_poscnt[16];

  const int base = blockIdx.x * 16;

  // Stage A rows into LDS (vector copies).
  {
    const v8h* src = (const v8h*)(enh + (size_t)base * D_DIM);
    v8h* dst = (v8h*)As;
    for (int i = threadIdx.x; i < 16 * D_DIM / 8; i += 256) dst[i] = src[i];
  }
  if (threadIdx.x < 16) {
    llab[threadIdx.x] = labels[base + threadIdx.x];
    s_sumexp[threadIdx.x] = 0.f;
    s_possum[threadIdx.x] = 0.f;
    s_poscnt[threadIdx.x] = 0.f;
  }
  __syncthreads();

  const int lane = threadIdx.x & 31;
  const int wid  = threadIdx.x >> 5;
  const int lh   = lane & 15;
  const int hi   = lane >> 4;

  float sumexp[8], possum[8], poscnt[8];
  int rlab[8], rowg[8];
#pragma unroll
  for (int v = 0; v < 8; v++) {
    sumexp[v] = 0.f; possum[v] = 0.f; poscnt[v] = 0.f;
    rlab[v] = llab[v + hi * 8];
    rowg[v] = base + v + hi * 8;
  }

  // A fragment source: row lh; lane half hi takes K [k0+16*hi, k0+16*hi+16), same order as B.
  const _Float16* arow = As + lh * D_DIM;
  const int c0 = wid * (B_ROWS / 8);              // 1024 columns per wave

  for (int ct = 0; ct < (B_ROWS / 8) / 32; ct++) {
    const int cA = c0 + ct * 32 + lh;             // tile 0 column for this lane
    const int cB = cA + 16;                       // tile 1 column
    const int clabA = labels[cA];
    const int clabB = labels[cB];
    // B fragment: lanes 0-15 hold K[k0..k0+16), lanes 16-31 K[k0+16..k0+32)
    const _Float16* browA = enh + (size_t)cA * D_DIM + hi * 16;
    const _Float16* browB = enh + (size_t)cB * D_DIM + hi * 16;

    v8f acc0 = {};
    v8f acc1 = {};
#pragma unroll 2
    for (int k0 = 0; k0 < D_DIM; k0 += 32) {
      v16h a = *(const v16h*)(arow + k0 + hi * 16);       // K [k0+16h, k0+16h+16)
      v16h b0 = *(const v16h*)(browA + k0);
      v16h b1 = *(const v16h*)(browB + k0);
      acc0 = __builtin_amdgcn_wmma_f32_16x16x32_f16(
          false, a, false, b0, (short)0, acc0, false, false);
      acc1 = __builtin_amdgcn_wmma_f32_16x16x32_f16(
          false, a, false, b1, (short)0, acc1, false, false);
    }

    // Branchless epilogue: C/D layout -> VGPR v holds row (v + hi*8), col = lh.
    // Selects lower to v_cndmask; EXEC stays all-ones (no divergence).
#pragma unroll
    for (int v = 0; v < 8; v++) {
      const float S0 = acc0[v] * TEMPR;
      const float S1 = acc1[v] * TEMPR;
      const float e0 = __expf(S0 - TEMPR);   // fixed-shift lse, S <= TEMPR
      const float e1 = __expf(S1 - TEMPR);
      const bool nd0 = (cA != rowg[v]);      // off-diagonal
      const bool nd1 = (cB != rowg[v]);
      const bool ps0 = nd0 && (clabA == rlab[v]);
      const bool ps1 = nd1 && (clabB == rlab[v]);
      sumexp[v] += (nd0 ? e0 : 0.f) + (nd1 ? e1 : 0.f);
      possum[v] += (ps0 ? S0 : 0.f) + (ps1 ? S1 : 0.f);
      poscnt[v] += (ps0 ? 1.f : 0.f) + (ps1 ? 1.f : 0.f);
    }
  }

  // Reduce each row's stats across its 16 column-lanes (xor masks stay in half).
#pragma unroll
  for (int v = 0; v < 8; v++) {
    for (int m = 8; m >= 1; m >>= 1) {
      sumexp[v] += __shfl_xor(sumexp[v], m, 32);
      possum[v] += __shfl_xor(possum[v], m, 32);
      poscnt[v] += __shfl_xor(poscnt[v], m, 32);
    }
  }
  if (lh == 0) {
#pragma unroll
    for (int v = 0; v < 8; v++) {
      const int ridx = v + hi * 8;
      atomicAdd(&s_sumexp[ridx], sumexp[v]);   // ds_add_f32 across 8 waves
      atomicAdd(&s_possum[ridx], possum[v]);
      atomicAdd(&s_poscnt[ridx], poscnt[v]);
    }
  }
  __syncthreads();
  if (threadIdx.x < 16) {
    const float lse = TEMPR + __logf(s_sumexp[threadIdx.x]);
    const float cnt = s_poscnt[threadIdx.x];
    const float inner = s_possum[threadIdx.x] - cnt * lse;
    contr_row[base + threadIdx.x] = (cnt > 0.5f) ? (-inner / cnt) : 0.f;
  }
}

// ---------------------------------------------------------------- kernel 4
__global__ __launch_bounds__(256) void finalize_kernel(
    const float* __restrict__ contr_row, const float* __restrict__ ce_row,
    float* __restrict__ out_loss) {
  __shared__ float w1[8], w2[8];
  float c1 = 0.f, c2 = 0.f;
  for (int i = threadIdx.x; i < B_ROWS; i += 256) {
    c1 += contr_row[i];
    c2 += ce_row[i];
  }
  for (int m = 16; m >= 1; m >>= 1) {
    c1 += __shfl_xor(c1, m, 32);
    c2 += __shfl_xor(c2, m, 32);
  }
  if ((threadIdx.x & 31) == 0) { w1[threadIdx.x >> 5] = c1; w2[threadIdx.x >> 5] = c2; }
  __syncthreads();
  if (threadIdx.x == 0) {
    float s1 = 0.f, s2 = 0.f;
    for (int i = 0; i < 8; i++) { s1 += w1[i]; s2 += w2[i]; }
    out_loss[0] = LAMBDA * s1 + (1.0f - LAMBDA) * (s2 / (float)B_ROWS);
  }
}

// ----------------------------------------------------------------
extern "C" void kernel_launch(void* const* d_in, const int* in_sizes, int n_in,
                              void* d_out, int out_size, void* d_ws, size_t ws_size,
                              hipStream_t stream) {
  (void)in_sizes; (void)n_in; (void)out_size; (void)ws_size;
  const float* cls    = (const float*)d_in[0];
  const float* pooled = (const float*)d_in[1];
  const int*   labels = (const int*)d_in[2];
  const float* W      = (const float*)d_in[3];
  const float* bias   = (const float*)d_in[4];
  float* out = (float*)d_out;           // [0] = loss, [1..] = logits (8192x7)

  char* ws = (char*)d_ws;
  _Float16* enh    = (_Float16*)ws;                             // 12,582,912 B
  float* ce_row    = (float*)(ws + (size_t)B_ROWS * D_DIM * 2); // 32 KB
  float* contr_row = ce_row + B_ROWS;                           // 32 KB

  normalize_f16_kernel<<<B_ROWS, 256, 0, stream>>>(cls, enh);
  logits_ce_kernel<<<B_ROWS, 256, 0, stream>>>(pooled, W, bias, labels, out + 1, ce_row);
  scl_gram_kernel<<<B_ROWS / 16, 256, 0, stream>>>(enh, labels, contr_row);
  finalize_kernel<<<1, 256, 0, stream>>>(contr_row, ce_row, out);
}
